// NumericHGN_21766894256453
// MI455X (gfx1250) — matrix operand order repaired and checked by reading, not compiled, not run on hardware
//
#include <hip/hip_runtime.h>
#include <hip/hip_bf16.h>

#define HID 256
#define HEADS 4
#define HEAD_DIM 64

typedef unsigned short u16;
typedef unsigned int u32;

typedef __attribute__((ext_vector_type(16))) __bf16 v16bf;
typedef __attribute__((ext_vector_type(8)))  float  v8f;

union Frag16 { v16bf v; uint4 q[2]; };

__device__ __forceinline__ u16 f2bf(float f) {
    union { float f; u32 u; } x; x.f = f;
    u32 u = x.u;
    u32 r = u + 0x7FFFu + ((u >> 16) & 1u);
    if ((u & 0x7F800000u) == 0x7F800000u) r = u;   // inf/nan passthrough
    return (u16)(r >> 16);
}
__device__ __forceinline__ float bf2f(u16 s) {
    union { u32 u; float f; } x; x.u = ((u32)s) << 16; return x.f;
}
__device__ __forceinline__ float wred32(float s) {
    #pragma unroll
    for (int m = 16; m > 0; m >>= 1) s += __shfl_xor(s, m, 32);
    return s;
}

// ---------------- zero init (atomically-accumulated buffers) ----------------
__global__ void zero_kernel(u32* __restrict__ p, size_t n) {
    size_t i = (size_t)blockIdx.x * blockDim.x + threadIdx.x;
    if (i < n) p[i] = 0u;
}

// ---------------- weight convert + transpose: W[K][NC] f32 -> Wt[NC][K] bf16 ----------------
// grid = (K, NC/256), block = 256: coalesced reads of one W row per block-row
__global__ void wconv_kernel(const float* __restrict__ W, u16* __restrict__ Wt, int K, int NC) {
    int k = blockIdx.x;
    int n = blockIdx.y * 256 + threadIdx.x;
    if (n >= NC) return;
    Wt[(size_t)n * K + k] = f2bf(W[(size_t)k * NC + n]);
}

// ---------------- layernorm: f32 [N,256] -> bf16 [N,256], one wave per node ----------------
__global__ void ln_kernel(const float* __restrict__ x, const float* __restrict__ g,
                          const float* __restrict__ b, u16* __restrict__ out, int N) {
    int node = blockIdx.x * (blockDim.x >> 5) + (threadIdx.x >> 5);
    int lane = threadIdx.x & 31;
    if (node >= N) return;
    const float4* row = (const float4*)(x + (size_t)node * HID);
    float4 v0 = row[lane * 2];
    float4 v1 = row[lane * 2 + 1];
    float s = v0.x + v0.y + v0.z + v0.w + v1.x + v1.y + v1.z + v1.w;
    float mu = wred32(s) * (1.0f / HID);
    float a0 = v0.x - mu, a1 = v0.y - mu, a2 = v0.z - mu, a3 = v0.w - mu;
    float a4 = v1.x - mu, a5 = v1.y - mu, a6 = v1.z - mu, a7 = v1.w - mu;
    float vs = a0*a0 + a1*a1 + a2*a2 + a3*a3 + a4*a4 + a5*a5 + a6*a6 + a7*a7;
    float var = wred32(vs) * (1.0f / HID);
    float rstd = rsqrtf(var + 1e-5f);
    int base = lane * 8;
    float av[8] = {a0,a1,a2,a3,a4,a5,a6,a7};
    union { u16 h[8]; uint4 q; } o;
    #pragma unroll
    for (int i = 0; i < 8; ++i)
        o.h[i] = f2bf(av[i] * rstd * g[base + i] + b[base + i]);
    ((uint4*)(out + (size_t)node * HID))[lane] = o.q;
}

// ---------------- f32 -> bf16 bulk convert ----------------
__global__ void cvt_f2bf_kernel(const float* __restrict__ in, u16* __restrict__ out, size_t n) {
    size_t i = ((size_t)blockIdx.x * blockDim.x + threadIdx.x) * 4;
    if (i >= n) return;
    float4 v = *(const float4*)(in + i);
    uint2 o;
    o.x = (u32)f2bf(v.x) | ((u32)f2bf(v.y) << 16);
    o.y = (u32)f2bf(v.z) | ((u32)f2bf(v.w) << 16);
    *(uint2*)(out + i) = o;
}

// ---------------- WMMA GEMM fragment helpers ----------------
__device__ __forceinline__ void load_a(Frag16& a, const u16* aRow, int kb) {
    a.q[0] = *(const uint4*)(aRow + kb);
    a.q[1] = *(const uint4*)(aRow + kb + 16);
}
__device__ __forceinline__ void load_b4(Frag16* b, const u16* bCol, int kb, int K) {
    #pragma unroll
    for (int t = 0; t < 4; ++t) {
        const u16* bp = bCol + (size_t)(t << 4) * K + kb;
        b[t].q[0] = *(const uint4*)(bp);
        b[t].q[1] = *(const uint4*)(bp + 16);
    }
}
__device__ __forceinline__ void wmma4(v8f* acc, const Frag16& a, const Frag16* b) {
    #pragma unroll
    for (int t = 0; t < 4; ++t)
        acc[t] = __builtin_amdgcn_wmma_f32_16x16x32_bf16(
            false, a.v, false, b[t].v, (short)0, acc[t], false, false);
}

// ---------------- WMMA GEMM: out[M,NC] = A[M,K](bf16) @ Wt[NC,K](bf16)^T + bias ----------------
// MODE 0: store bf16 ; MODE 1: store f32 (+ residual) ; MODE 2: relu, store bf16
// Requires M % 16 == 0, K % 64 == 0, NC % 256 == 0. Block = 4 waves, 16 rows x 256 cols.
template<int MODE>
__global__ __launch_bounds__(128) void wmma_gemm(
    const u16* __restrict__ A, const u16* __restrict__ Wt,
    const float* __restrict__ bias, const float* __restrict__ resid,
    void* __restrict__ outp, int K, int NC)
{
    const int lane    = threadIdx.x & 31;
    const int wave    = threadIdx.x >> 5;
    const int rowBase = blockIdx.y << 4;
    const int colWave = (blockIdx.x << 8) + (wave << 6);
    const int nIdx    = lane & 15;
    const int hi      = lane >> 4;

    // A fragment (16x32 bf16): lane row = nIdx; K chunks [hi*8, hi*8+8) and [hi*8+16, hi*8+24)
    const u16* aRow = A + (size_t)(rowBase + nIdx) * K + (hi << 3);
    // B fragment (32x16 bf16): lane col = nIdx; K chunks [hi*8, hi*8+8) and [hi*8+16, hi*8+24)
    const u16* bCol = Wt + (size_t)(colWave + nIdx) * K + (hi << 3);

    v8f acc[4] = {v8f{0.f}, v8f{0.f}, v8f{0.f}, v8f{0.f}};

    // software-pipelined, double-buffered over pairs of k-steps (K multiple of 64)
    Frag16 aA, bA[4], aB, bB[4];
    load_a(aA, aRow, 0);   load_b4(bA, bCol, 0, K);
    load_a(aB, aRow, 32);  load_b4(bB, bCol, 32, K);
    int kb = 0;
    for (; kb + 64 < K; kb += 64) {
        wmma4(acc, aA, bA);
        load_a(aA, aRow, kb + 64);  load_b4(bA, bCol, kb + 64, K);
        wmma4(acc, aB, bB);
        load_a(aB, aRow, kb + 96);  load_b4(bB, bCol, kb + 96, K);
    }
    wmma4(acc, aA, bA);
    wmma4(acc, aB, bB);

    #pragma unroll
    for (int t = 0; t < 4; ++t) {
        const int col = colWave + (t << 4) + nIdx;
        const float bv = bias[col];
        #pragma unroll
        for (int r = 0; r < 8; ++r) {
            const int row = rowBase + (hi << 3) + r;
            const size_t o = (size_t)row * NC + col;
            float v = acc[t][r] + bv;
            if (MODE == 0)      ((u16*)outp)[o] = f2bf(v);
            else if (MODE == 1) ((float*)outp)[o] = resid[o] + v;
            else                ((u16*)outp)[o] = f2bf(v > 0.f ? v : 0.f);
        }
    }
}

// ---------------- edge logits + segment max (wave per edge) ----------------
__device__ __forceinline__ float dot2(u32 q, u32 k) {
    return bf2f((u16)(q & 0xffffu)) * bf2f((u16)(k & 0xffffu)) +
           bf2f((u16)(q >> 16))     * bf2f((u16)(k >> 16));
}
__global__ void edge_logits_kernel(const u16* __restrict__ Qb, const u16* __restrict__ Kb,
                                   const int* __restrict__ src, const int* __restrict__ dst,
                                   float* __restrict__ logits, u32* __restrict__ segmax, int E) {
    int e = blockIdx.x * (blockDim.x >> 5) + (threadIdx.x >> 5);
    int lane = threadIdx.x & 31;
    if (e >= E) return;
    int s = src[e], d = dst[e];
    uint4 qv = ((const uint4*)(Qb + (size_t)d * HID))[lane];
    uint4 kv = ((const uint4*)(Kb + (size_t)s * HID))[lane];
    float dot = dot2(qv.x, kv.x) + dot2(qv.y, kv.y) + dot2(qv.z, kv.z) + dot2(qv.w, kv.w);
    #pragma unroll
    for (int m = 1; m < 8; m <<= 1) dot += __shfl_xor(dot, m, 32);
    if ((lane & 7) == 0) {
        int h = lane >> 3;
        float lg = dot * 0.125f;  // 1/sqrt(HEAD_DIM)
        logits[(size_t)e * HEADS + h] = lg;
        u32 bits = __float_as_uint(lg);
        u32 key = (bits & 0x80000000u) ? ~bits : (bits | 0x80000000u);
        atomicMax(&segmax[(size_t)d * HEADS + h], key);
    }
}

// ---------------- exp(l - max) + segment sum (thread per edge-head) ----------------
__global__ void edge_exp_kernel(float* __restrict__ logits, const int* __restrict__ dst,
                                const u32* __restrict__ segmax, float* __restrict__ segsum, int E4) {
    int idx = blockIdx.x * blockDim.x + threadIdx.x;
    if (idx >= E4) return;
    int e = idx >> 2, h = idx & 3;
    int d = dst[e];
    u32 key = segmax[(size_t)d * HEADS + h];
    u32 bits = (key & 0x80000000u) ? (key & 0x7FFFFFFFu) : ~key;
    float mx = __uint_as_float(bits);
    float p = __expf(logits[idx] - mx);
    logits[idx] = p;
    unsafeAtomicAdd(&segsum[(size_t)d * HEADS + h], p);
}

// ---------------- normalize + scatter V (wave per edge) ----------------
__global__ void edge_scatter_kernel(const u16* __restrict__ Vb, const int* __restrict__ src,
                                    const int* __restrict__ dst, const float* __restrict__ p,
                                    const float* __restrict__ segsum, float* __restrict__ hnew, int E) {
    int e = blockIdx.x * (blockDim.x >> 5) + (threadIdx.x >> 5);
    int lane = threadIdx.x & 31;
    if (e >= E) return;
    int s = src[e], d = dst[e];
    int h = lane >> 3;
    float w = p[(size_t)e * HEADS + h] / segsum[(size_t)d * HEADS + h];
    uint4 vv = ((const uint4*)(Vb + (size_t)s * HID))[lane];
    float* outp = hnew + (size_t)d * HID + lane * 8;
    float vals[8];
    vals[0] = bf2f((u16)(vv.x & 0xffffu)); vals[1] = bf2f((u16)(vv.x >> 16));
    vals[2] = bf2f((u16)(vv.y & 0xffffu)); vals[3] = bf2f((u16)(vv.y >> 16));
    vals[4] = bf2f((u16)(vv.z & 0xffffu)); vals[5] = bf2f((u16)(vv.z >> 16));
    vals[6] = bf2f((u16)(vv.w & 0xffffu)); vals[7] = bf2f((u16)(vv.w >> 16));
    #pragma unroll
    for (int i = 0; i < 8; ++i) unsafeAtomicAdd(outp + i, vals[i] * w);
}

extern "C" void kernel_launch(void* const* d_in, const int* in_sizes, int n_in,
                              void* d_out, int out_size, void* d_ws, size_t ws_size,
                              hipStream_t stream) {
    const float* h   = (const float*)d_in[0];
    const int*   src = (const int*)d_in[1];
    const int*   dst = (const int*)d_in[2];
    const float* Wq  = (const float*)d_in[3];  const float* bq = (const float*)d_in[4];
    const float* Wk  = (const float*)d_in[5];  const float* bk = (const float*)d_in[6];
    const float* Wv  = (const float*)d_in[7];  const float* bv = (const float*)d_in[8];
    const float* Wo  = (const float*)d_in[9];  const float* bo = (const float*)d_in[10];
    const float* W1  = (const float*)d_in[11]; const float* b1 = (const float*)d_in[12];
    const float* W2  = (const float*)d_in[13]; const float* b2 = (const float*)d_in[14];
    const float* g1  = (const float*)d_in[15]; const float* be1 = (const float*)d_in[16];
    const float* g2  = (const float*)d_in[17]; const float* be2 = (const float*)d_in[18];

    const int N = in_sizes[0] / HID;   // 100000 (multiple of 16)
    const int E = in_sizes[1];         // 1600000

    // ---- workspace layout (256B-aligned slabs) ----
    char* ws = (char*)d_ws;
    size_t off = 0;
    auto alloc = [&](size_t bytes) -> char* {
        char* pp = ws + off;
        off += (bytes + 255) & ~(size_t)255;
        return pp;
    };
    u16*  hn     = (u16*)alloc((size_t)N * HID * 2);          // LN1 output, bf16
    u16*  Qb     = (u16*)alloc((size_t)N * HID * 2);
    u16*  Kb     = (u16*)alloc((size_t)N * HID * 2);
    u16*  Vb     = (u16*)alloc((size_t)N * HID * 2);          // Kb,Vb contiguous
    u16*  WqT    = (u16*)alloc((size_t)HID * HID * 2);
    u16*  WkT    = (u16*)alloc((size_t)HID * HID * 2);
    u16*  WvT    = (u16*)alloc((size_t)HID * HID * 2);
    u16*  WoT    = (u16*)alloc((size_t)HID * HID * 2);
    u16*  W1T    = (u16*)alloc((size_t)HID * 2 * HID * 2);
    u16*  W2T    = (u16*)alloc((size_t)2 * HID * HID * 2);
    float* logits = (float*)alloc((size_t)E * HEADS * 4);
    u32*  segmax = (u32*)alloc((size_t)N * HEADS * 4);
    float* segsum = (float*)alloc((size_t)N * HEADS * 4);     // contiguous with segmax
    float* hnew   = (float*)alloc((size_t)N * HID * 4);       // contiguous with segsum
    // aliases (lifetimes don't overlap):
    u16*  hn2  = hn;            // bf16(hnew), after QKV GEMMs consumed hn
    float* h1  = hnew;          // h + attn_out, after hnew converted to hn2
    u16*  h2n  = Qb;            // LN2 output, after edge phase consumed Qb
    u16*  ffn1 = Kb;            // [N,512] bf16, spans Kb+Vb

    // 1) weight convert + transpose
    { dim3 g(HID, 1);        wconv_kernel<<<g, 256, 0, stream>>>(Wq, WqT, HID, HID); }
    { dim3 g(HID, 1);        wconv_kernel<<<g, 256, 0, stream>>>(Wk, WkT, HID, HID); }
    { dim3 g(HID, 1);        wconv_kernel<<<g, 256, 0, stream>>>(Wv, WvT, HID, HID); }
    { dim3 g(HID, 1);        wconv_kernel<<<g, 256, 0, stream>>>(Wo, WoT, HID, HID); }
    { dim3 g(HID, 2);        wconv_kernel<<<g, 256, 0, stream>>>(W1, W1T, HID, 2*HID); }
    { dim3 g(2*HID, 1);      wconv_kernel<<<g, 256, 0, stream>>>(W2, W2T, 2*HID, HID); }

    // 2) LN1
    ln_kernel<<<(N + 7) / 8, 256, 0, stream>>>(h, g1, be1, hn, N);

    // 3) zero segmax+segsum+hnew (contiguous)
    size_t zn = (size_t)N * (HEADS + HEADS + HID);
    zero_kernel<<<(unsigned)((zn + 255) / 256), 256, 0, stream>>>(segmax, zn);

    // 4) QKV GEMMs -> bf16
    dim3 gB(1, N / 16);
    wmma_gemm<0><<<gB, 128, 0, stream>>>(hn, WqT, bq, nullptr, Qb, HID, HID);
    wmma_gemm<0><<<gB, 128, 0, stream>>>(hn, WkT, bk, nullptr, Kb, HID, HID);
    wmma_gemm<0><<<gB, 128, 0, stream>>>(hn, WvT, bv, nullptr, Vb, HID, HID);

    // 5) edge attention
    edge_logits_kernel<<<(E + 7) / 8, 256, 0, stream>>>(Qb, Kb, src, dst, logits, segmax, E);
    edge_exp_kernel<<<(E * HEADS + 255) / 256, 256, 0, stream>>>(logits, dst, segmax, segsum, E * HEADS);
    edge_scatter_kernel<<<(E + 7) / 8, 256, 0, stream>>>(Vb, src, dst, logits, segsum, hnew, E);

    // 6) aggregate -> bf16, Wo GEMM with residual h -> h1 (f32)
    size_t nh = (size_t)N * HID;
    cvt_f2bf_kernel<<<(unsigned)((nh / 4 + 255) / 256), 256, 0, stream>>>(hnew, hn2, nh);
    wmma_gemm<1><<<gB, 128, 0, stream>>>(hn2, WoT, bo, h, h1, HID, HID);

    // 7) LN2, FFN, final residual -> d_out
    ln_kernel<<<(N + 7) / 8, 256, 0, stream>>>(h1, g2, be2, h2n, N);
    dim3 gF(2, N / 16);
    wmma_gemm<2><<<gF, 128, 0, stream>>>(h2n, W1T, b1, nullptr, ffn1, HID, 2*HID);
    wmma_gemm<1><<<gB, 128, 0, stream>>>(ffn1, W2T, b2, h1, d_out, 2*HID, HID);
}
